// DenseLanguageGuidance_60919816127010
// MI455X (gfx1250) — matrix operand order repaired and checked by reading, not compiled, not run on hardware
//
#include <hip/hip_runtime.h>

// ---------------------------------------------------------------------------
// DenseLanguageGuidance cross-attention for MI455X (gfx1250, wave32, WMMA).
// - All GEMMs: v_wmma_f32_16x16x32_bf16, f32 accumulate, NT layout so both
//   operands are K-contiguous -> per-lane b128 loads.
// - Big GEMMs stage the block-shared B panel into LDS with the Tensor Data
//   Mover (tensor_load_to_lds + s_wait_tensorcnt), freeing the VMEM pipe for
//   streaming the A operand.
// ---------------------------------------------------------------------------

typedef __bf16 bf16;
typedef __attribute__((ext_vector_type(16))) __bf16 v16bf;
typedef __attribute__((ext_vector_type(8)))  __bf16 v8bf;
typedef __attribute__((ext_vector_type(8)))  float  v8f;
typedef __attribute__((ext_vector_type(4)))  unsigned int u32x4;
typedef __attribute__((ext_vector_type(8)))  int i32x8;
typedef __attribute__((ext_vector_type(4)))  int i32x4;

#define C_DIM  1024
#define B_DIM  8
#define HW_DIM 4096
#define N_TOK  77
#define N_PAD  96   // language tokens padded to multiple of 32 for K-tiling

// ---- WMMA fragment loads (bf16, K-contiguous memory) ----------------------
// A 16x32: lanes 0-15 row M=lane: K=[k0..k0+7] in v0-3, K=[k0+8..k0+15] in v4-7
//          lanes 16-31 row M=lane-16: K=[k0+16..23] / K=[k0+24..31]
__device__ __forceinline__ v16bf load_frag_a(const bf16* __restrict__ A, int ldk,
                                             int m0, int k0, int lane) {
  const bf16* p = A + (long)(m0 + (lane & 15)) * ldk + (k0 + ((lane >> 4) << 4));
  v8bf lo = *(const v8bf*)(p);
  v8bf hi = *(const v8bf*)(p + 8);
  v16bf f;
#pragma unroll
  for (int i = 0; i < 8; ++i) { f[i] = lo[i]; f[i + 8] = hi[i]; }
  return f;
}

// B 32x16: lane holds column N=lane&15, K-run of 16 starting k0+(lane>>4)*16.
__device__ __forceinline__ v16bf load_frag_b(const bf16* __restrict__ Bt, int ldk,
                                             int n0, int k0, int lane) {
  const bf16* p = Bt + (long)(n0 + (lane & 15)) * ldk + (k0 + ((lane >> 4) << 4));
  return *(const v16bf*)(p);
}

__device__ __forceinline__ v8f wmma_bf16(v16bf a, v16bf b, v8f c) {
  return __builtin_amdgcn_wmma_f32_16x16x32_bf16(false, a, false, b, (short)0, c,
                                                 false, false);
}

// ---- Tensor Data Mover: 2D tile (rows x k_len bf16) global -> LDS ---------
// D# built per CDNA5 ISA. Tile start = tensor
// origin; dim0 stride may exceed tensor_dim0 (interior tile). data_size=2B.
// 6-arg builtin form (clang-23 / therock-10.0 lane): groups 2/3 + extra
// int32x8 group are unused for a plain 2D tile -> zero-filled.
__device__ __forceinline__ void tdm_load_2d(bf16* lds_dst, const bf16* gsrc,
                                            int k_len, int rows, int row_stride) {
  unsigned long long ga = (unsigned long long)gsrc;
  unsigned lds = (unsigned)(unsigned long long)lds_dst;  // low 32 bits = LDS addr
  u32x4 g0;
  g0[0] = 1u;                                    // count=1, user-mode, no gather
  g0[1] = lds;                                   // lds_addr
  g0[2] = (unsigned)(ga & 0xffffffffu);          // global_addr[31:0]
  g0[3] = (unsigned)((ga >> 32) & 0x01ffffffu) | (2u << 30);  // addr[56:32]|type=2
  i32x8 g1;
  g1[0] = (int)(1u << 16);                       // data_size=1 (2 bytes)
  g1[1] = (int)(((unsigned)k_len & 0xffffu) << 16);            // tensor_dim0 lo
  g1[2] = (int)((((unsigned)k_len >> 16) & 0xffffu) |
                (((unsigned)rows & 0xffffu) << 16));           // td0 hi | td1 lo
  g1[3] = (int)(((((unsigned)rows >> 16) & 0xffffu)) |
                (((unsigned)k_len & 0xffffu) << 16));          // td1 hi | tile_dim0
  g1[4] = (int)((unsigned)rows & 0xffffu);       // tile_dim1 | tile_dim2=0
  g1[5] = (int)(unsigned)row_stride;             // tensor_dim0_stride[31:0]
  g1[6] = 0;
  g1[7] = 0;
  i32x4 z4 = {0, 0, 0, 0};
  i32x8 z8 = {0, 0, 0, 0, 0, 0, 0, 0};
  __builtin_amdgcn_tensor_load_to_lds(g0, g1, z4, z4, z8, 0);
}

// ---- Generic batched NT GEMM: D = A * Bt^T (+bias), bf16 in, f32 acc ------
// Block = 256 threads = 8 waves (BWY x BWX); wave computes WM x WN 16x16
// tiles. ld(A)=ld(Bt)=K, ld(D)=N. blockIdx.z = batch. When TDM_ is set the
// block's B panel (NB rows x 256-wide K chunk) is DMA'd to LDS by the TDM.
template <int WM, int WN, int BWY, int BWX, bool BIAS, bool TDM_, typename OutT>
__global__ void __launch_bounds__(256)
gemm_nt(const bf16* __restrict__ A, const bf16* __restrict__ Bt,
        const float* __restrict__ bias, OutT* __restrict__ D,
        int K, int N, long strA, long strB, long strD) {
  extern __shared__ __align__(32) bf16 bpanel[];  // [NB][KC] when TDM_
  const int lane = threadIdx.x & 31;
  const int wid  = threadIdx.x >> 5;
  const int wy   = wid / BWX;
  const int wx   = wid % BWX;
  const bf16* Ab = A + (long)blockIdx.z * strA;
  const bf16* Bb = Bt + (long)blockIdx.z * strB;
  OutT*       Db = D + (long)blockIdx.z * strD;
  const int NB = 16 * WN * BWX;                   // B panel rows per block
  const int nblock0 = blockIdx.x * NB;
  const int m0 = (blockIdx.y * BWY + wy) * (16 * WM);
  const int n0 = nblock0 + wx * (16 * WN);
  const int KC = TDM_ ? (K >= 256 ? 256 : K) : K; // K chunk staged per DMA

  v8f acc[WM][WN] = {};
  for (int kc = 0; kc < K; kc += KC) {
    if (TDM_) {
      __syncthreads();                            // previous chunk consumed
      if (wid == 0) {
        tdm_load_2d(bpanel, Bb + (long)nblock0 * K + kc, KC, NB, K);
        __builtin_amdgcn_s_wait_tensorcnt(0);
      }
      __syncthreads();                            // panel visible to all waves
    }
    for (int k0 = 0; k0 < KC; k0 += 32) {
      v16bf af[WM], bfm[WN];
#pragma unroll
      for (int i = 0; i < WM; ++i)
        af[i] = load_frag_a(Ab, K, m0 + 16 * i, kc + k0, lane);
#pragma unroll
      for (int j = 0; j < WN; ++j)
        bfm[j] = TDM_ ? load_frag_b(bpanel, KC, (wx * 16 * WN) + 16 * j, k0, lane)
                      : load_frag_b(Bb, K, n0 + 16 * j, kc + k0, lane);
#pragma unroll
      for (int i = 0; i < WM; ++i)
#pragma unroll
        for (int j = 0; j < WN; ++j) acc[i][j] = wmma_bf16(af[i], bfm[j], acc[i][j]);
    }
  }

  // D layout: VGPR r <-> (m = r + 8*(lane>>4), n = lane&15)
  const int mrow = (lane >> 4) << 3;
  const int ncol = lane & 15;
#pragma unroll
  for (int j = 0; j < WN; ++j) {
    const int n = n0 + 16 * j + ncol;
    const float bv = BIAS ? bias[n] : 0.0f;
#pragma unroll
    for (int i = 0; i < WM; ++i)
#pragma unroll
      for (int r = 0; r < 8; ++r) {
        float v = acc[i][j][r] + bv;
        Db[(long)(m0 + 16 * i + mrow + r) * N + n] = (OutT)v;
      }
  }
}

// ---- Fused logits + masked softmax ----------------------------------------
__global__ void __launch_bounds__(256)
attn_softmax(const bf16* __restrict__ vk, const bf16* __restrict__ lk,
             bf16* __restrict__ attn) {
  const int lane = threadIdx.x & 31;
  const int wid  = threadIdx.x >> 5;
  const int b    = blockIdx.y;
  const int m0   = (blockIdx.x * 8 + wid) * 16;
  const bf16* Ab = vk + (long)b * HW_DIM * C_DIM;
  const bf16* Bb = lk + (long)b * N_PAD * C_DIM;

  v8f acc[5] = {};
  for (int k0 = 0; k0 < C_DIM; k0 += 32) {
    v16bf a = load_frag_a(Ab, C_DIM, m0, k0, lane);
#pragma unroll
    for (int nt = 0; nt < 5; ++nt) {
      v16bf bf_ = load_frag_b(Bb, C_DIM, nt * 16, k0, lane);
      acc[nt] = wmma_bf16(a, bf_, acc[nt]);
    }
  }

  const float scale = 0.03125f;  // 1024^-0.5
  const int ncol = lane & 15;
  const int mrow = (lane >> 4) << 3;
  bf16* dst_base = attn + ((long)b * HW_DIM + m0 + mrow) * N_PAD;
#pragma unroll
  for (int r = 0; r < 8; ++r) {
    float v[5];
    float mx = -3.0e38f;
#pragma unroll
    for (int nt = 0; nt < 5; ++nt) {
      int n = nt * 16 + ncol;
      v[nt] = (n < N_TOK) ? acc[nt][r] * scale : -3.0e38f;
      mx = fmaxf(mx, v[nt]);
    }
#pragma unroll
    for (int o = 1; o < 16; o <<= 1) mx = fmaxf(mx, __shfl_xor(mx, o, 32));
    float p[5], s = 0.0f;
#pragma unroll
    for (int nt = 0; nt < 5; ++nt) {
      p[nt] = (v[nt] > -1.0e38f) ? __expf(v[nt] - mx) : 0.0f;
      s += p[nt];
    }
#pragma unroll
    for (int o = 1; o < 16; o <<= 1) s += __shfl_xor(s, o, 32);
    const float inv = 1.0f / s;
    bf16* dst = dst_base + (long)r * N_PAD;
#pragma unroll
    for (int nt = 0; nt < 5; ++nt) dst[nt * 16 + ncol] = (bf16)(p[nt] * inv);
    dst[80 + ncol] = (bf16)0.0f;  // exact-zero pad columns 80..95
  }
}

// ---- LDS-tiled batched transpose: in (R x Cc) -> out (Cc x R), bf16 -------
__global__ void __launch_bounds__(256)
transpose_bf16(const bf16* __restrict__ in, bf16* __restrict__ out,
               int R, int Cc, long strI, long strO) {
  __shared__ bf16 tile[32][33];
  const bf16* ib = in + (long)blockIdx.z * strI;
  bf16*       ob = out + (long)blockIdx.z * strO;
  const int c0 = blockIdx.x * 32;
  const int r0 = blockIdx.y * 32;
  const int tx = threadIdx.x & 31;
  const int ty = threadIdx.x >> 5;
#pragma unroll
  for (int i = 0; i < 32; i += 8)
    tile[ty + i][tx] = ib[(long)(r0 + ty + i) * Cc + c0 + tx];
  __syncthreads();
#pragma unroll
  for (int i = 0; i < 32; i += 8)
    ob[(long)(c0 + ty + i) * R + r0 + tx] = tile[tx][ty + i];
}

// ---- Elementwise helpers --------------------------------------------------
__global__ void cast_f32_bf16(const float* __restrict__ in, bf16* __restrict__ out,
                              long n) {
  long i = (long)blockIdx.x * blockDim.x + threadIdx.x;
  if (i < n) out[i] = (bf16)in[i];
}

__global__ void pad_lang_bf16(const float* __restrict__ in, bf16* __restrict__ out) {
  long i = (long)blockIdx.x * blockDim.x + threadIdx.x;  // over B*96*C
  long c = i % C_DIM;
  long n = (i / C_DIM) % N_PAD;
  long b = i / ((long)C_DIM * N_PAD);
  out[i] = (n < N_TOK) ? (bf16)in[(b * N_TOK + n) * C_DIM + c] : (bf16)0.0f;
}

__global__ void add_bf16(const bf16* __restrict__ a, const bf16* __restrict__ b,
                         bf16* __restrict__ o, long n) {
  long i = (long)blockIdx.x * blockDim.x + threadIdx.x;
  if (i < n) o[i] = (bf16)((float)a[i] + (float)b[i]);
}

// ---------------------------------------------------------------------------
extern "C" void kernel_launch(void* const* d_in, const int* in_sizes, int n_in,
                              void* d_out, int out_size, void* d_ws, size_t ws_size,
                              hipStream_t stream) {
  (void)in_sizes; (void)n_in; (void)out_size; (void)ws_size;
  const float* vis  = (const float*)d_in[0];
  const float* lang = (const float*)d_in[1];
  const float* W_vk = (const float*)d_in[2];
  const float* b_vk = (const float*)d_in[3];
  const float* W_lk = (const float*)d_in[4];
  const float* b_lk = (const float*)d_in[5];
  const float* W_vv = (const float*)d_in[6];
  const float* b_vv = (const float*)d_in[7];
  const float* W_lv = (const float*)d_in[8];
  const float* b_lv = (const float*)d_in[9];
  float* out = (float*)d_out;

  const long MV = (long)B_DIM * HW_DIM;  // 32768 vision rows
  const long ML = (long)B_DIM * N_PAD;   // 768 padded language rows

  size_t off = 0;
  auto alloc = [&](long elems) -> bf16* {
    bf16* p = (bf16*)((char*)d_ws + off);
    off += ((size_t)elems * sizeof(bf16) + 255) & ~(size_t)255;
    return p;
  };
  bf16* visX  = alloc(MV * C_DIM);
  bf16* langX = alloc(ML * C_DIM);
  bf16* Wvk   = alloc((long)C_DIM * C_DIM);
  bf16* Wlk   = alloc((long)C_DIM * C_DIM);
  bf16* Wvv   = alloc((long)C_DIM * C_DIM);
  bf16* Wlv   = alloc((long)C_DIM * C_DIM);
  bf16* vk    = alloc(MV * C_DIM);                    // (B*HW, C)
  bf16* vv    = alloc(MV * C_DIM);                    // (B*HW, C)
  bf16* lk    = alloc(ML * C_DIM);                    // (B, 96, C)
  bf16* lv    = alloc(ML * C_DIM);                    // (B, 96, C)
  bf16* lvT   = alloc((long)B_DIM * C_DIM * N_PAD);   // (B, C, 96)
  bf16* vvT   = alloc((long)B_DIM * C_DIM * HW_DIM);  // (B, C, HW)
  bf16* attn  = alloc((long)B_DIM * HW_DIM * N_PAD);  // (B, HW, 96)
  bf16* attnT = alloc((long)B_DIM * N_PAD * HW_DIM);  // (B, 96, HW)
  bf16* VT    = alloc((long)B_DIM * C_DIM * N_PAD);   // (B, C, 96)
  bf16* SUMT  = alloc((long)B_DIM * C_DIM * N_PAD);   // (B, C, 96)

  const dim3 blk(256);
  const long WN_ = (long)C_DIM * C_DIM;
  const size_t shProj  = (size_t)64 * 256 * sizeof(bf16);   // NB=64, KC=256
  const size_t shFinal = (size_t)64 * N_PAD * sizeof(bf16); // NB=64, KC=96

  // 1) bf16 casts / padding
  cast_f32_bf16<<<(MV * C_DIM + 255) / 256, blk, 0, stream>>>(vis, visX, MV * C_DIM);
  cast_f32_bf16<<<(WN_ + 255) / 256, blk, 0, stream>>>(W_vk, Wvk, WN_);
  cast_f32_bf16<<<(WN_ + 255) / 256, blk, 0, stream>>>(W_lk, Wlk, WN_);
  cast_f32_bf16<<<(WN_ + 255) / 256, blk, 0, stream>>>(W_vv, Wvv, WN_);
  cast_f32_bf16<<<(WN_ + 255) / 256, blk, 0, stream>>>(W_lv, Wlv, WN_);
  pad_lang_bf16<<<(ML * C_DIM + 255) / 256, blk, 0, stream>>>(lang, langX);

  // 2) projections: Y = X @ W^T + b (block 256x64, wave 64x32, TDM B panels)
  gemm_nt<4, 2, 4, 2, true, true, bf16>
      <<<dim3(C_DIM / 64, MV / 256, 1), blk, shProj, stream>>>(
          visX, Wvk, b_vk, vk, C_DIM, C_DIM, 0, 0, 0);
  gemm_nt<4, 2, 4, 2, true, true, bf16>
      <<<dim3(C_DIM / 64, MV / 256, 1), blk, shProj, stream>>>(
          visX, Wvv, b_vv, vv, C_DIM, C_DIM, 0, 0, 0);
  gemm_nt<4, 2, 4, 2, true, true, bf16>
      <<<dim3(C_DIM / 64, ML / 256, 1), blk, shProj, stream>>>(
          langX, Wlk, b_lk, lk, C_DIM, C_DIM, 0, 0, 0);
  gemm_nt<4, 2, 4, 2, true, true, bf16>
      <<<dim3(C_DIM / 64, ML / 256, 1), blk, shProj, stream>>>(
          langX, Wlv, b_lv, lv, C_DIM, C_DIM, 0, 0, 0);

  // 3) transposes for K-contiguous operands
  transpose_bf16<<<dim3(C_DIM / 32, N_PAD / 32, B_DIM), blk, 0, stream>>>(
      lv, lvT, N_PAD, C_DIM, (long)N_PAD * C_DIM, (long)C_DIM * N_PAD);
  transpose_bf16<<<dim3(C_DIM / 32, HW_DIM / 32, B_DIM), blk, 0, stream>>>(
      vv, vvT, HW_DIM, C_DIM, (long)HW_DIM * C_DIM, (long)C_DIM * HW_DIM);

  // 4) logits + softmax -> attn (B,HW,96), exact zeros for n>=77
  attn_softmax<<<dim3(HW_DIM / 16 / 8, B_DIM), blk, 0, stream>>>(vk, lk, attn);
  transpose_bf16<<<dim3(N_PAD / 32, HW_DIM / 32, B_DIM), blk, 0, stream>>>(
      attn, attnT, HW_DIM, N_PAD, (long)HW_DIM * N_PAD, (long)N_PAD * HW_DIM);

  // 5) V^T[c,n] = sum_s vvT[c,s] * attnT[n,s]   (K = 4096, direct loads)
  gemm_nt<2, 1, 4, 2, false, false, bf16>
      <<<dim3(N_PAD / 32, C_DIM / 128, B_DIM), blk, 0, stream>>>(
          vvT, attnT, nullptr, VT, HW_DIM, N_PAD,
          (long)C_DIM * HW_DIM, (long)N_PAD * HW_DIM, (long)C_DIM * N_PAD);

  // 6) SUMT = lvT + VT  (fuses lang_attended + projected into one GEMM)
  {
    long n = (long)B_DIM * C_DIM * N_PAD;
    add_bf16<<<(n + 255) / 256, blk, 0, stream>>>(lvT, VT, SUMT, n);
  }

  // 7) out[s,c] = sum_n attn[s,n] * SUMT[c,n]   (K = 96, f32 output, TDM)
  gemm_nt<4, 2, 4, 2, false, true, float>
      <<<dim3(C_DIM / 64, HW_DIM / 256, B_DIM), blk, shFinal, stream>>>(
          attn, SUMT, nullptr, out, N_PAD, C_DIM,
          (long)HW_DIM * N_PAD, (long)C_DIM * N_PAD, (long)HW_DIM * C_DIM);
}
